// Gai_HGNN_17978733101720
// MI455X (gfx1250) — matrix operand order repaired and checked by reading, not compiled, not run on hardware
//
#include <hip/hip_runtime.h>
#include <hip/hip_bf16.h>

typedef __attribute__((ext_vector_type(16))) _Float16 v16h;
typedef __attribute__((ext_vector_type(8)))  float    v8f;

#define DEV static __device__ __forceinline__
#define NEGBIG (-1.0e9f)

// ---------------- wave32 reductions ----------------
DEV float redsum32(float v) {
#pragma unroll
  for (int off = 16; off >= 1; off >>= 1) v += __shfl_xor(v, off);
  return v;
}
DEV float redsum16(float v) {
#pragma unroll
  for (int off = 8; off >= 1; off >>= 1) v += __shfl_xor(v, off);
  return v;
}
DEV float redmax16(float v) {
#pragma unroll
  for (int off = 8; off >= 1; off >>= 1) v = fmaxf(v, __shfl_xor(v, off));
  return v;
}

// ---------------- WMMA fragment helpers (gfx1250 wave32 layouts) ----------------
// 16-bit A matrix 16x32: lane L: M = L&15, h = L>>4.
// element e (v16h): K = 16*h + e (contiguous, matches B fragments)
DEV int a_kpat(int e, int h) {
  int v = e >> 1;
  (void)v;
  return (h << 4) + e;
}
DEV v16h afrag_f32(const float* rowp, int h) {
  v16h a;
#pragma unroll
  for (int e = 0; e < 16; ++e) a[e] = (_Float16)rowp[a_kpat(e, h)];
  return a;
}
DEV v16h afrag_lds_f16(const _Float16* rowp, int h) {
  v16h a;
#pragma unroll
  for (int e = 0; e < 16; ++e) a[e] = rowp[a_kpat(e, h)];
  return a;
}
// 16-bit B matrix 32x16: lane L: N = L&15, h = L>>4; element e -> K = 16*h + e.
// Per-lane data is a contiguous 16-element K-run of the source row.
DEV v16h bfrag_seq_f32(const float* p) {
  v16h b;
#pragma unroll
  for (int e = 0; e < 16; ++e) b[e] = (_Float16)p[e];
  return b;
}
DEV v16h bfrag_seq_f16(const _Float16* p) {
  v16h b;
#pragma unroll
  for (int e = 0; e < 16; ++e) b[e] = p[e];
  return b;
}
DEV v16h bfrag_strided_f16(const _Float16* base, int h, int stride, int col) {
  v16h b;
#pragma unroll
  for (int e = 0; e < 16; ++e) b[e] = base[(16 * h + e) * stride + col];
  return b;
}
DEV v8f zero8() {
  v8f c;
#pragma unroll
  for (int i = 0; i < 8; ++i) c[i] = 0.0f;
  return c;
}
DEV v8f wmma16(v16h a, v16h b, v8f c) {
  // D(f32 16x16) = A(f16 16x32) x B(f16 32x16) + C
  return __builtin_amdgcn_wmma_f32_16x16x32_f16(false, a, false, b, (short)0, c,
                                                false, false);
}

// ---------------- K1: x = l2norm_caps(X @ W + b); writes 3 copies ----------------
__global__ void proj_norm_kernel(const float* __restrict__ X, const float* __restrict__ W,
                                 const float* __restrict__ b, float* __restrict__ xo,
                                 float* __restrict__ co, float* __restrict__ c2o, int F) {
  __shared__ float row[640];
  const int r = blockIdx.x;
  const int t = threadIdx.x;  // 64 threads: wave0 = capsule0, wave1 = capsule1
  for (int f = t; f < F; f += 64) row[f] = X[(size_t)r * F + f];
  __syncthreads();
  float acc = b[t];
  for (int f = 0; f < F; ++f) acc = fmaf(row[f], W[f * 64 + t], acc);
  const float ss = redsum32(acc * acc);
  const float v = acc / fmaxf(sqrtf(ss), 1e-12f);
  const size_t o = (size_t)r * 64 + t;
  xo[o] = v; co[o] = v; c2o[o] = v;
}

// ---------------- K3: one routing iteration over edges (wave per edge) ----------------
__global__ void routing_edge_kernel(const float* __restrict__ x, const float* __restrict__ c,
                                    float* __restrict__ cnxt, const int* __restrict__ src,
                                    const int* __restrict__ trg, int E) {
  const int lane = threadIdx.x & 31;
  const int wave = (blockIdx.x * blockDim.x + threadIdx.x) >> 5;
  const int nwaves = (gridDim.x * blockDim.x) >> 5;
  for (int e = wave; e < E; e += nwaves) {
    const int s = src[e], t = trg[e];
    const float z0 = x[s * 64 + lane];
    const float z1 = x[s * 64 + 32 + lane];
    const float t0 = c[t * 64 + lane];
    const float t1 = c[t * 64 + 32 + lane];
    const float d0 = redsum32(z0 * t0);   // capsule-0 dot
    const float d1 = redsum32(z1 * t1);   // capsule-1 dot
    const float mx = fmaxf(d0, d1);
    const float e0 = __expf(d0 - mx), e1 = __expf(d1 - mx);
    const float inv = 1.0f / (e0 + e1);
    atomicAdd(&cnxt[t * 64 + lane], e0 * inv * z0);
    atomicAdd(&cnxt[t * 64 + 32 + lane], e1 * inv * z1);
  }
}

// ---------------- K4: per-capsule L2 norm, writes two destinations ----------------
__global__ void caps_norm_kernel(const float* __restrict__ src, float* __restrict__ d0,
                                 float* __restrict__ d1, int n) {
  const int r = blockIdx.x * 4 + (threadIdx.x >> 6);
  const int t = threadIdx.x & 63;
  if (r >= n) return;
  const size_t o = (size_t)r * 64 + t;
  float v = src[o];
  const float ss = redsum32(v * v);
  v = v / fmaxf(sqrtf(ss), 1e-12f);
  d0[o] = v; d1[o] = v;
}

// ---------------- K5: attention fuse over 3 views ----------------
__global__ void attn_fuse_kernel(const float* __restrict__ z0p, const float* __restrict__ z1p,
                                 const float* __restrict__ z2p, const float* __restrict__ W1,
                                 const float* __restrict__ b1, const float* __restrict__ W2,
                                 float* __restrict__ out) {
  __shared__ float z[3][64];
  __shared__ float partial[3][2];
  const int r = blockIdx.x;
  const int t = threadIdx.x;  // 64
  const size_t o = (size_t)r * 64 + t;
  z[0][t] = z0p[o]; z[1][t] = z1p[o]; z[2][t] = z2p[o];
  __syncthreads();
  const int wid = t >> 5;
#pragma unroll
  for (int s = 0; s < 3; ++s) {
    float h = b1[t];
    for (int k = 0; k < 64; ++k) h = fmaf(z[s][k], W1[k * 64 + t], h);
    h = (h > 0.0f) ? h : 0.2f * h;  // leaky relu 0.2
    const float p = redsum32(h * W2[t]);
    if ((t & 31) == 0) partial[s][wid] = p;
  }
  __syncthreads();
  const float w0 = partial[0][0] + partial[0][1];
  const float w1 = partial[1][0] + partial[1][1];
  const float w2 = partial[2][0] + partial[2][1];
  const float mx = fmaxf(w0, fmaxf(w1, w2));
  const float e0 = __expf(w0 - mx), e1 = __expf(w1 - mx), e2 = __expf(w2 - mx);
  const float inv = 1.0f / (e0 + e1 + e2);
  const float be0 = e0 * inv + 1e-6f, be1 = e1 * inv + 1e-6f, be2 = e2 * inv + 1e-6f;
  const float binv = 1.0f / (be0 + be1 + be2);
  out[o] = (be0 * z[0][t] + be1 * z[1][t] + be2 * z[2][t]) * binv;
}

// ---------------- K6: relu(X @ W + b), D=64 ----------------
__global__ void linrelu_kernel(const float* __restrict__ X, const float* __restrict__ W,
                               const float* __restrict__ b, float* __restrict__ out) {
  __shared__ float row[64];
  const int r = blockIdx.x;
  const int t = threadIdx.x;
  row[t] = X[(size_t)r * 64 + t];
  __syncthreads();
  float acc = b[t];
  for (int k = 0; k < 64; ++k) acc = fmaf(row[k], W[k * 64 + t], acc);
  out[(size_t)r * 64 + t] = fmaxf(acc, 0.0f);
}

// ---------------- K7a: masked-softmax stats (row max & exp-sum), WMMA scores ----------------
// One wave owns 16 Q-rows; loops all Kt rows in tiles of 16; S tile via 2 WMMA (K=64).
__global__ void attn_stats_kernel(const float* __restrict__ Q, const float* __restrict__ Kt,
                                  const float* __restrict__ Am, long mrs, long mcs,
                                  int Ncols, float* __restrict__ mOut,
                                  float* __restrict__ lOut) {
  const int wid = threadIdx.x >> 5, lane = threadIdx.x & 31;
  const int h = lane >> 4, nn = lane & 15;
  const int i0 = (blockIdx.x * 8 + wid) * 16;
  const float* qrow = Q + (size_t)(i0 + nn) * 64;
  const v16h aq0 = afrag_f32(qrow, h);
  const v16h aq1 = afrag_f32(qrow + 32, h);
  float mln[8], lln[8];
#pragma unroll
  for (int r = 0; r < 8; ++r) { mln[r] = -3.0e38f; lln[r] = 0.0f; }
  for (int j0 = 0; j0 < Ncols; j0 += 16) {
    const float* krow = Kt + (size_t)(j0 + nn) * 64 + h * 16;
    const v16h b0 = bfrag_seq_f32(krow);
    const v16h b1 = bfrag_seq_f32(krow + 32);
    v8f c = zero8();
    c = wmma16(aq0, b0, c);
    c = wmma16(aq1, b1, c);
#pragma unroll
    for (int r = 0; r < 8; ++r) {   // per-lane online max/sum (cols n, n+16, ...)
      const long i = i0 + r + 8 * h;
      const float mv = Am[i * mrs + (long)(j0 + nn) * mcs];
      const float s = (mv > 0.0f) ? c[r] : NEGBIG;
      const float mnew = fmaxf(mln[r], s);
      lln[r] = lln[r] * __expf(mln[r] - mnew) + __expf(s - mnew);
      mln[r] = mnew;
    }
  }
#pragma unroll
  for (int r = 0; r < 8; ++r) {     // merge 16 lanes of a group
    const float mrow = redmax16(mln[r]);
    const float lrow = redsum16(lln[r] * __expf(mln[r] - mrow));
    if (nn == 0) { mOut[i0 + r + 8 * h] = mrow; lOut[i0 + r + 8 * h] = lrow; }
  }
}

// ---------------- K7b: Out = Tpre + softmax(S) @ Kt, flash-style, WMMA both GEMMs ----------
__global__ void attn_apply_kernel(const float* __restrict__ Q, const float* __restrict__ Kt,
                                  const float* __restrict__ Am, long mrs, long mcs,
                                  int Ncols, const float* __restrict__ mIn,
                                  const float* __restrict__ lIn,
                                  const float* __restrict__ Tpre, float* __restrict__ Out) {
  __shared__ _Float16 ich[32 * 64];      // Kt chunk (32 rows x 64 dims), block-shared
  __shared__ _Float16 pt[8][16 * 32];    // per-wave P tile (D-layout -> A-layout transpose)
  const int wid = threadIdx.x >> 5, lane = threadIdx.x & 31;
  const int h = lane >> 4, nn = lane & 15;
  const int i0 = (blockIdx.x * 8 + wid) * 16;
  const float* qrow = Q + (size_t)(i0 + nn) * 64;
  const v16h aq0 = afrag_f32(qrow, h);
  const v16h aq1 = afrag_f32(qrow + 32, h);
  float mrow[8];
#pragma unroll
  for (int r = 0; r < 8; ++r) mrow[r] = mIn[i0 + r + 8 * h];
  v8f o0 = zero8(), o1 = zero8(), o2 = zero8(), o3 = zero8();
  for (int j0 = 0; j0 < Ncols; j0 += 32) {
    __syncthreads();
    for (int idx = threadIdx.x; idx < 32 * 64; idx += 256)
      ich[idx] = (_Float16)Kt[(size_t)(j0 + (idx >> 6)) * 64 + (idx & 63)];
    __syncthreads();
#pragma unroll
    for (int sub = 0; sub < 2; ++sub) {  // two 16-col S tiles per chunk
      const _Float16* kr = &ich[(sub * 16 + nn) * 64 + h * 16];
      const v16h b0 = bfrag_seq_f16(kr);
      const v16h b1 = bfrag_seq_f16(kr + 32);
      v8f c = zero8();
      c = wmma16(aq0, b0, c);
      c = wmma16(aq1, b1, c);
#pragma unroll
      for (int r = 0; r < 8; ++r) {
        const long i = i0 + r + 8 * h;
        const long j = j0 + sub * 16 + nn;
        const float mv = Am[i * mrs + j * mcs];
        const float s = (mv > 0.0f) ? c[r] : NEGBIG;
        pt[wid][(r + 8 * h) * 32 + sub * 16 + nn] = (_Float16)__expf(s - mrow[r]);
      }
    }
    // P (16x32) as A-frag from LDS; Kt chunk as B (32 x 64) in 4 N-tiles.
    const v16h ap = afrag_lds_f16(&pt[wid][nn * 32], h);
    o0 = wmma16(ap, bfrag_strided_f16(ich, h, 64, 0 + nn), o0);
    o1 = wmma16(ap, bfrag_strided_f16(ich, h, 64, 16 + nn), o1);
    o2 = wmma16(ap, bfrag_strided_f16(ich, h, 64, 32 + nn), o2);
    o3 = wmma16(ap, bfrag_strided_f16(ich, h, 64, 48 + nn), o3);
  }
  float linv[8];
#pragma unroll
  for (int r = 0; r < 8; ++r) linv[r] = 1.0f / lIn[i0 + r + 8 * h];
#pragma unroll
  for (int r = 0; r < 8; ++r) {
    const size_t i = i0 + r + 8 * h;
    Out[i * 64 + 0 + nn]  = Tpre[i * 64 + 0 + nn]  + o0[r] * linv[r];
    Out[i * 64 + 16 + nn] = Tpre[i * 64 + 16 + nn] + o1[r] * linv[r];
    Out[i * 64 + 32 + nn] = Tpre[i * 64 + 32 + nn] + o2[r] * linv[r];
    Out[i * 64 + 48 + nn] = Tpre[i * 64 + 48 + nn] + o3[r] * linv[r];
  }
}

// ---------------- K8: Out[MxN] = A[Mx64] @ B[Nx64]^T (final decoder) ----------------
__global__ void gemm_nt_kernel(const float* __restrict__ A, const float* __restrict__ B,
                               float* __restrict__ Out, int M, int N) {
  const int wid = threadIdx.x >> 5, lane = threadIdx.x & 31;
  const int h = lane >> 4, nn = lane & 15;
  const long wg = (long)blockIdx.x * 8 + wid;
  const int tiles_n = N >> 4;
  const int tm = (int)(wg / tiles_n), tn = (int)(wg % tiles_n);
  if (tm * 16 >= M) return;
  const int i0 = tm * 16, j0 = tn * 16;
  const float* arow = A + (size_t)(i0 + nn) * 64;
  const float* brow = B + (size_t)(j0 + nn) * 64 + h * 16;
  v8f c = zero8();
  c = wmma16(afrag_f32(arow, h), bfrag_seq_f32(brow), c);
  c = wmma16(afrag_f32(arow + 32, h), bfrag_seq_f32(brow + 32), c);
#pragma unroll
  for (int r = 0; r < 8; ++r)
    Out[(size_t)(i0 + r + 8 * h) * N + j0 + nn] = c[r];
}

// =======================================================================================
extern "C" void kernel_launch(void* const* d_in, const int* in_sizes, int n_in,
                              void* d_out, int out_size, void* d_ws, size_t ws_size,
                              hipStream_t stream) {
  (void)in_sizes; (void)n_in; (void)out_size; (void)ws_size;
  constexpr int N_NC = 10240, N_D = 5120, F_NC = 599, F_D = 91;
  constexpr int E_NC = 327680, E_D = 163840, Dm = 64;

  const float* NC[3] = {(const float*)d_in[0], (const float*)d_in[1], (const float*)d_in[2]};
  const float* DX[3] = {(const float*)d_in[3], (const float*)d_in[4], (const float*)d_in[5]};
  const int*  ENC[3] = {(const int*)d_in[6], (const int*)d_in[7], (const int*)d_in[8]};
  const int*  ED[3]  = {(const int*)d_in[9], (const int*)d_in[10], (const int*)d_in[11]};
  const float* A    = (const float*)d_in[14];
  const float* W_nc = (const float*)d_in[16]; const float* b_nc = (const float*)d_in[17];
  const float* W_d  = (const float*)d_in[18]; const float* b_d  = (const float*)d_in[19];
  const float* aW1  = (const float*)d_in[20]; const float* ab1  = (const float*)d_in[21];
  const float* aW2  = (const float*)d_in[22];
  const float* jW   = (const float*)d_in[23]; const float* jb   = (const float*)d_in[24];
  float* out = (float*)d_out;

  float* ws = (float*)d_ws;
  size_t off = 0;
  auto alloc = [&](size_t n) { float* p = ws + off; off += n; return p; };
  float* x_init = alloc((size_t)N_NC * Dm);
  float* c_cur  = alloc((size_t)N_NC * Dm);
  float* c_nxt  = alloc((size_t)N_NC * Dm);
  float* encNC[3]; for (int g = 0; g < 3; ++g) encNC[g] = alloc((size_t)N_NC * Dm);
  float* encD[3];  for (int g = 0; g < 3; ++g) encD[g]  = alloc((size_t)N_D * Dm);
  float* nc_f  = alloc((size_t)N_NC * Dm);
  float* d_f   = alloc((size_t)N_D * Dm);
  float* tu    = alloc((size_t)N_NC * Dm);
  float* ti    = alloc((size_t)N_D * Dm);
  float* upd_u = alloc((size_t)N_NC * Dm);
  float* upd_i = alloc((size_t)N_D * Dm);
  float* m_u = alloc(N_NC); float* l_u = alloc(N_NC);
  float* m_i = alloc(N_D);  float* l_i = alloc(N_D);

  auto run_encoder = [&](const float* X, int n, int F, const float* W, const float* b,
                         const int* ei, int E, float* outp) {
    proj_norm_kernel<<<n, 64, 0, stream>>>(X, W, b, x_init, c_cur, c_nxt, F);
    for (int it = 0; it < 3; ++it) {
      routing_edge_kernel<<<2048, 256, 0, stream>>>(x_init, c_cur, c_nxt, ei, ei + E, E);
      caps_norm_kernel<<<(n + 3) / 4, 256, 0, stream>>>(c_nxt, (it == 2) ? outp : c_cur,
                                                        c_nxt, n);
    }
  };
  for (int g = 0; g < 3; ++g) run_encoder(NC[g], N_NC, F_NC, W_nc, b_nc, ENC[g], E_NC, encNC[g]);
  for (int g = 0; g < 3; ++g) run_encoder(DX[g], N_D, F_D, W_d, b_d, ED[g], E_D, encD[g]);

  attn_fuse_kernel<<<N_NC, 64, 0, stream>>>(encNC[0], encNC[1], encNC[2], aW1, ab1, aW2, nc_f);
  attn_fuse_kernel<<<N_D, 64, 0, stream>>>(encD[0], encD[1], encD[2], aW1, ab1, aW2, d_f);

  linrelu_kernel<<<N_NC, 64, 0, stream>>>(nc_f, jW, jb, tu);
  linrelu_kernel<<<N_D, 64, 0, stream>>>(d_f, jW, jb, ti);

  // user side: mask = A[i,j] (row stride N_D, col stride 1)
  attn_stats_kernel<<<N_NC / 128, 256, 0, stream>>>(nc_f, d_f, A, (long)N_D, 1L, N_D, m_u, l_u);
  attn_apply_kernel<<<N_NC / 128, 256, 0, stream>>>(nc_f, d_f, A, (long)N_D, 1L, N_D,
                                                    m_u, l_u, tu, upd_u);
  // item side: mask = A[j,i] = A.T (row stride 1, col stride N_D)
  attn_stats_kernel<<<N_D / 128, 256, 0, stream>>>(d_f, nc_f, A, 1L, (long)N_D, N_NC, m_i, l_i);
  attn_apply_kernel<<<N_D / 128, 256, 0, stream>>>(d_f, nc_f, A, 1L, (long)N_D, N_NC,
                                                   m_i, l_i, ti, upd_i);

  // final: out[5120,10240] = upd_item @ upd_user^T
  gemm_nt_kernel<<<(N_D / 16) * (N_NC / 16) / 8, 256, 0, stream>>>(upd_i, upd_u, out, N_D, N_NC);
}
